// WeatherTransformer_28063316312424
// MI455X (gfx1250) — matrix operand order repaired and checked by reading, not compiled, not run on hardware
//
#include <hip/hip_runtime.h>
#include <stdint.h>
#include <stddef.h>

// ---------------- problem dims ----------------
#define B_    512
#define S_    336
#define F_    12
#define PATCH_ 6
#define D_    512
#define H_    8
#define NL_   6
#define DFF_  2048
#define HOR_  24
#define NP_   56
#define DH_   64
#define ROWS_ (B_ * NP_)   // 28672 token rows

// ---------------- CDNA5 WMMA types ----------------
typedef __attribute__((ext_vector_type(16))) __bf16 v16bf;
typedef __attribute__((ext_vector_type(8)))  float  v8f;

union Frag { uint4 q[2]; v16bf v; };
static_assert(sizeof(Frag) == 32, "frag is 8 VGPRs");

__device__ __forceinline__ unsigned short f2bf(float f) {
  unsigned int u = __float_as_uint(f);
  u += 0x7FFFu + ((u >> 16) & 1u);          // round-to-nearest-even
  return (unsigned short)(u >> 16);
}

// gfx1250 async memory->LDS copy (ASYNCcnt-tracked). LDS offset = low 32 bits
// of the generic pointer (ISA 10.2: flat LDS addr = {aperture, offset32}).
__device__ __forceinline__ void async_copy_b128(const void* gptr, void* lptr) {
  const unsigned lds = (unsigned)(size_t)lptr;
  asm volatile("global_load_async_to_lds_b128 %0, %1, off"
               :: "v"(lds), "v"(gptr) : "memory");
}
__device__ __forceinline__ void wait_async_le(int n) {
  if (n == 0) asm volatile("s_wait_asynccnt 0" ::: "memory");
  else        asm volatile("s_wait_asynccnt 4" ::: "memory");
}

// =====================================================================
// Generic GEMM: C[M,N] = A[M,K](bf16 rowmajor) * B, with B given as
// Bt[N,K] (bf16, transposed) so B-fragments are contiguous-K loads.
// EPI 0: out_bf16 = acc + bias
// EPI 1: R(f32)  += acc + bias          (residual in place)
// EPI 2: out_bf16 = gelu(acc + bias)    (tanh approximation, as jax.nn.gelu)
// Tile: 128x128, K-step 32, 256 threads = 8 waves; wave => 16 rows x 128 cols
// (8 WMMAs per A-fragment). Double-buffered LDS filled by async-to-LDS.
// =====================================================================
template <int EPI>
__global__ __launch_bounds__(256, 1)
void gemm_bf16_wmma(const unsigned short* __restrict__ A,
                    const unsigned short* __restrict__ Bt,
                    const float* __restrict__ bias,
                    float* __restrict__ R,
                    unsigned short* __restrict__ O,
                    int M, int N, int K) {
  __shared__ __align__(16) unsigned short As[2][128][40];  // stride 80B: conflict-free frags
  __shared__ __align__(16) unsigned short Bs[2][128][40];

  const int tid  = threadIdx.x;
  const int wave = tid >> 5;
  const int lane = tid & 31;
  const int g    = lane >> 4;   // half-wave group
  const int n    = lane & 15;   // row-within-frag / column index
  const int m0   = blockIdx.x * 128;
  const int n0   = blockIdx.y * 128;

  // Stage one 128x32 A tile + 128x32 B tile: 512+512 16B slots, 4 async
  // instructions per thread (ASYNCcnt += 4 per wave per tile).
  auto stage = [&](int buf, int kk) {
#pragma unroll
    for (int i = 0; i < 2; ++i) {
      const int slot = tid + 256 * i;
      const int r = slot >> 2, sg = slot & 3;
      async_copy_b128(A  + (size_t)(m0 + r) * K + kk + sg * 8, &As[buf][r][sg * 8]);
      async_copy_b128(Bt + (size_t)(n0 + r) * K + kk + sg * 8, &Bs[buf][r][sg * 8]);
    }
  };

  v8f acc[8] = {};
  stage(0, 0);
  int cur = 0;
  for (int kk = 0; kk < K; kk += 32) {
    if (kk + 32 < K) {
      stage(cur ^ 1, kk + 32);   // prefetch next tile into other buffer
      wait_async_le(4);          // oldest 4 (current tile) complete, in order
    } else {
      wait_async_le(0);
    }
    __syncthreads();             // make all waves' async fills visible

    // A fragment (K = 16g + e, matching B fragment)
    Frag a;
    a.q[0] = *(const uint4*)&As[cur][wave * 16 + n][16 * g];
    a.q[1] = *(const uint4*)&As[cur][wave * 16 + n][16 * g + 8];
#pragma unroll
    for (int t = 0; t < 8; ++t) {
      // B fragment: column n of B = row (t*16+n) of Bt, K = 16g + e
      Frag b;
      b.q[0] = *(const uint4*)&Bs[cur][t * 16 + n][16 * g];
      b.q[1] = *(const uint4*)&Bs[cur][t * 16 + n][16 * g + 8];
      acc[t] = __builtin_amdgcn_wmma_f32_16x16x32_bf16(
          false, a.v, false, b.v, (short)0, acc[t], false, false);
    }
    __syncthreads();             // all waves done reading buf[cur]
    cur ^= 1;
  }

  // ---- epilogue: D layout => element (M = v + 8g + 16*wave, N = 16t + n) ----
#pragma unroll
  for (int t = 0; t < 8; ++t) {
#pragma unroll
    for (int v = 0; v < 8; ++v) {
      const int gm = m0 + wave * 16 + v + 8 * g;
      const int gn = n0 + t * 16 + n;
      float val = acc[t][v] + bias[gn];
      if (EPI == 0) {
        O[(size_t)gm * N + gn] = f2bf(val);
      } else if (EPI == 1) {
        R[(size_t)gm * N + gn] += val;
      } else {
        const float c = 0.7978845608028654f * (val + 0.044715f * val * val * val);
        O[(size_t)gm * N + gn] = f2bf(0.5f * val * (1.0f + tanhf(c)));
      }
    }
  }
}

// =====================================================================
// Attention: one block per (b, head). 128 threads = 4 waves.
// S = Q K^T * 1/8 + ALiBi, masked softmax over 56 keys (padded to 64),
// O = P V.  Q/K rows & V columns staged in LDS so all WMMA fragments are
// contiguous 16B loads.
// =====================================================================
__global__ __launch_bounds__(128, 1)
void attention_wmma(const unsigned short* __restrict__ Q,
                    const unsigned short* __restrict__ Kc,
                    const unsigned short* __restrict__ V,
                    unsigned short* __restrict__ Obuf) {
  __shared__ __align__(16) unsigned short Qs[64][72];  // stride 144B, conflict-free
  __shared__ __align__(16) unsigned short Ks[64][72];
  __shared__ __align__(16) unsigned short Vt[64][72];  // V transposed: [dh][key]
  __shared__ __align__(16) unsigned short Ps[64][72];  // softmax probs (bf16)

  const int tid = threadIdx.x;
  const int b = blockIdx.x >> 3;
  const int h = blockIdx.x & 7;
  const size_t base = ((size_t)b * NP_) * D_ + (size_t)h * DH_;

  // ---- load Q, K rows and V transposed; zero-pad rows 56..63 ----
  {
    const int row = tid >> 1, half = tid & 1;
    if (row < NP_) {
#pragma unroll
      for (int j = 0; j < 4; ++j) {
        const size_t off = base + (size_t)row * D_ + half * 32 + j * 8;
        *(uint4*)&Qs[row][half * 32 + j * 8] = *(const uint4*)(Q + off);
        *(uint4*)&Ks[row][half * 32 + j * 8] = *(const uint4*)(Kc + off);
        union { uint4 q; unsigned short s[8]; } d;
        d.q = *(const uint4*)(V + off);
#pragma unroll
        for (int e = 0; e < 8; ++e) Vt[half * 32 + j * 8 + e][row] = d.s[e];
      }
    } else {
      const uint4 z = make_uint4(0, 0, 0, 0);
#pragma unroll
      for (int j = 0; j < 4; ++j) {
        *(uint4*)&Qs[row][half * 32 + j * 8] = z;
        *(uint4*)&Ks[row][half * 32 + j * 8] = z;
#pragma unroll
        for (int e = 0; e < 8; ++e) Vt[half * 32 + j * 8 + e][row] = 0;
      }
    }
  }
  __syncthreads();

  const int wave = tid >> 5;
  const int lane = tid & 31;
  const int g = lane >> 4;
  const int n = lane & 15;

  // ---- S = Q K^T : wave handles 16-query strip x all 64 keys ----
  v8f sa[4] = {};
#pragma unroll
  for (int ks = 0; ks < 64; ks += 32) {
    Frag a;
    a.q[0] = *(const uint4*)&Qs[wave * 16 + n][ks + 16 * g];
    a.q[1] = *(const uint4*)&Qs[wave * 16 + n][ks + 16 * g + 8];
#pragma unroll
    for (int t = 0; t < 4; ++t) {
      Frag bb;
      bb.q[0] = *(const uint4*)&Ks[t * 16 + n][ks + 16 * g];
      bb.q[1] = *(const uint4*)&Ks[t * 16 + n][ks + 16 * g + 8];
      sa[t] = __builtin_amdgcn_wmma_f32_16x16x32_bf16(
          false, a.v, false, bb.v, (short)0, sa[t], false, false);
    }
  }

  // ---- softmax with ALiBi; row lives on one half-wave (16 lanes x 4 accs) ----
  const float slope = exp2f(-(float)(h + 1));  // 2^(-8(h+1)/H), H=8
#pragma unroll
  for (int v = 0; v < 8; ++v) {
    const int qi = wave * 16 + v + 8 * g;
    float sv[4];
    float mx = -1e30f;
#pragma unroll
    for (int t = 0; t < 4; ++t) {
      const int kj = t * 16 + n;
      sv[t] = sa[t][v] * 0.125f - slope * fabsf((float)(qi - kj));
      if (kj < NP_) mx = fmaxf(mx, sv[t]);
    }
#pragma unroll
    for (int m = 1; m < 16; m <<= 1) mx = fmaxf(mx, __shfl_xor(mx, m, 32));
    float p[4], sum = 0.f;
#pragma unroll
    for (int t = 0; t < 4; ++t) {
      const int kj = t * 16 + n;
      p[t] = (kj < NP_) ? __expf(sv[t] - mx) : 0.f;
      sum += p[t];
    }
#pragma unroll
    for (int m = 1; m < 16; m <<= 1) sum += __shfl_xor(sum, m, 32);
    const float inv = (qi < NP_) ? (1.f / sum) : 0.f;  // padded queries -> P=0
#pragma unroll
    for (int t = 0; t < 4; ++t) Ps[qi][t * 16 + n] = f2bf(p[t] * inv);
  }
  __syncthreads();

  // ---- O = P V  (B-frag column dh = row of Vt, contiguous in key) ----
  v8f oa[4] = {};
#pragma unroll
  for (int ks = 0; ks < 64; ks += 32) {
    Frag a;
    a.q[0] = *(const uint4*)&Ps[wave * 16 + n][ks + 16 * g];
    a.q[1] = *(const uint4*)&Ps[wave * 16 + n][ks + 16 * g + 8];
#pragma unroll
    for (int t = 0; t < 4; ++t) {
      Frag bb;
      bb.q[0] = *(const uint4*)&Vt[t * 16 + n][ks + 16 * g];
      bb.q[1] = *(const uint4*)&Vt[t * 16 + n][ks + 16 * g + 8];
      oa[t] = __builtin_amdgcn_wmma_f32_16x16x32_bf16(
          false, a.v, false, bb.v, (short)0, oa[t], false, false);
    }
  }
#pragma unroll
  for (int v = 0; v < 8; ++v) {
    const int qi = wave * 16 + v + 8 * g;
    if (qi < NP_) {
#pragma unroll
      for (int t = 0; t < 4; ++t) {
        const int dh = t * 16 + n;
        Obuf[base + (size_t)qi * D_ + dh] = f2bf(oa[t][v]);
      }
    }
  }
}

// =====================================================================
// LayerNorm over D=512, output bf16. One wave per row, 8 rows per block.
// =====================================================================
__global__ __launch_bounds__(256, 1)
void layernorm_to_bf16(const float* __restrict__ X, const float* __restrict__ gam,
                       const float* __restrict__ bet, unsigned short* __restrict__ Y) {
  const int row  = blockIdx.x * 8 + (threadIdx.x >> 5);
  const int lane = threadIdx.x & 31;
  const float* x = X + (size_t)row * D_;

  float4 v[4];
  float s = 0.f;
#pragma unroll
  for (int c = 0; c < 4; ++c) {
    v[c] = *(const float4*)(x + lane * 4 + c * 128);
    s += v[c].x + v[c].y + v[c].z + v[c].w;
  }
#pragma unroll
  for (int m = 1; m < 32; m <<= 1) s += __shfl_xor(s, m, 32);
  const float mean = s * (1.f / 512.f);

  float q = 0.f;
#pragma unroll
  for (int c = 0; c < 4; ++c) {
    float a0 = v[c].x - mean, a1 = v[c].y - mean, a2 = v[c].z - mean, a3 = v[c].w - mean;
    q += a0 * a0 + a1 * a1 + a2 * a2 + a3 * a3;
  }
#pragma unroll
  for (int m = 1; m < 32; m <<= 1) q += __shfl_xor(q, m, 32);
  const float rstd = rsqrtf(q * (1.f / 512.f) + 1e-5f);

#pragma unroll
  for (int c = 0; c < 4; ++c) {
    const int col = lane * 4 + c * 128;
    const float4 gg = *(const float4*)(gam + col);
    const float4 bb = *(const float4*)(bet + col);
    union { unsigned short s[4]; uint2 u; } o;
    o.s[0] = f2bf((v[c].x - mean) * rstd * gg.x + bb.x);
    o.s[1] = f2bf((v[c].y - mean) * rstd * gg.y + bb.y);
    o.s[2] = f2bf((v[c].z - mean) * rstd * gg.z + bb.z);
    o.s[3] = f2bf((v[c].w - mean) * rstd * gg.w + bb.w);
    *(uint2*)(Y + (size_t)row * D_ + col) = o.u;
  }
}

// =====================================================================
// Patch embedding: h[r, :] = x_flat[r, 0:72] @ patch_W + patch_b + pos[r%56].
// (reshape B,S,F -> B,NP,72 is contiguous.) K=72 is tiny -> scalar FMA.
// =====================================================================
__global__ __launch_bounds__(256, 1)
void patch_embed(const float* __restrict__ x, const float* __restrict__ W,
                 const float* __restrict__ pb, const float* __restrict__ pos,
                 float* __restrict__ hout) {
  __shared__ float xs[72];
  const int r = blockIdx.x;
  const int p = r % NP_;
  const int t = threadIdx.x;
  if (t < 72) xs[t] = x[(size_t)r * 72 + t];
  __syncthreads();
#pragma unroll
  for (int i = 0; i < 2; ++i) {
    const int d = t + i * 256;
    float acc = pb[d] + pos[(size_t)p * D_ + d];
    for (int k = 0; k < 72; ++k) acc = fmaf(xs[k], W[(size_t)k * D_ + d], acc);
    hout[(size_t)r * D_ + d] = acc;
  }
}

// =====================================================================
// Weight prep: f32 [K][N] -> bf16 [N][K] (transpose+convert). 32x32 tiles.
// =====================================================================
__global__ __launch_bounds__(256, 1)
void transpose_f32_to_bf16(const float* __restrict__ in, unsigned short* __restrict__ out,
                           int K, int N) {
  __shared__ float tile[32][33];
  const int tx = threadIdx.x & 31, ty = threadIdx.x >> 5;
  const int n0 = blockIdx.x * 32, k0 = blockIdx.y * 32;
#pragma unroll
  for (int j = 0; j < 32; j += 8)
    tile[ty + j][tx] = in[(size_t)(k0 + ty + j) * N + n0 + tx];
  __syncthreads();
#pragma unroll
  for (int j = 0; j < 32; j += 8)
    out[(size_t)(n0 + ty + j) * K + k0 + tx] = f2bf(tile[tx][ty + j]);
}

// =====================================================================
// Mean-pool over NP, final LayerNorm, head projection (512 x 24). 1 block/b.
// =====================================================================
__global__ __launch_bounds__(256, 1)
void pool_ln_head(const float* __restrict__ hbuf, const float* __restrict__ fg,
                  const float* __restrict__ fb, const float* __restrict__ hW,
                  const float* __restrict__ hb, float* __restrict__ out) {
  __shared__ float pl[512];
  __shared__ float red[256];
  const int b = blockIdx.x, t = threadIdx.x;

#pragma unroll
  for (int i = 0; i < 2; ++i) {
    const int d = t + i * 256;
    float s = 0.f;
    for (int np = 0; np < NP_; ++np) s += hbuf[((size_t)b * NP_ + np) * D_ + d];
    pl[d] = s * (1.f / (float)NP_);
  }
  __syncthreads();

  red[t] = pl[t] + pl[t + 256];
  __syncthreads();
  for (int st = 128; st > 0; st >>= 1) {
    if (t < st) red[t] += red[t + st];
    __syncthreads();
  }
  const float mean = red[0] * (1.f / 512.f);
  __syncthreads();
  {
    const float a = pl[t] - mean, c = pl[t + 256] - mean;
    red[t] = a * a + c * c;
  }
  __syncthreads();
  for (int st = 128; st > 0; st >>= 1) {
    if (t < st) red[t] += red[t + st];
    __syncthreads();
  }
  const float rstd = rsqrtf(red[0] * (1.f / 512.f) + 1e-5f);
  __syncthreads();
#pragma unroll
  for (int i = 0; i < 2; ++i) {
    const int d = t + i * 256;
    pl[d] = (pl[d] - mean) * rstd * fg[d] + fb[d];
  }
  __syncthreads();

  if (t < HOR_ * 8) {  // 24 outputs x 8 partial dots of 64
    const int o = t >> 3, part = t & 7;
    float s = 0.f;
    for (int k = part * 64; k < part * 64 + 64; ++k) s += pl[k] * hW[(size_t)k * HOR_ + o];
    red[t] = s;
  }
  __syncthreads();
  if (t < HOR_) {
    float s = hb[t];
#pragma unroll
    for (int p = 0; p < 8; ++p) s += red[t * 8 + p];
    out[(size_t)b * HOR_ + t] = s;
  }
}

// =====================================================================
// Host-side orchestration
// =====================================================================
extern "C" void kernel_launch(void* const* d_in, const int* in_sizes, int n_in,
                              void* d_out, int out_size, void* d_ws, size_t ws_size,
                              hipStream_t stream) {
  (void)in_sizes; (void)n_in; (void)out_size; (void)ws_size;

  const float* x       = (const float*)d_in[0];
  const float* patch_W = (const float*)d_in[1];
  const float* patch_b = (const float*)d_in[2];
  const float* pos     = (const float*)d_in[3];
  const float* ln1_g   = (const float*)d_in[4];
  const float* ln1_b   = (const float*)d_in[5];
  const float* Wq = (const float*)d_in[6];  const float* bq = (const float*)d_in[7];
  const float* Wk = (const float*)d_in[8];  const float* bk = (const float*)d_in[9];
  const float* Wv = (const float*)d_in[10]; const float* bv = (const float*)d_in[11];
  const float* Wo = (const float*)d_in[12]; const float* bo = (const float*)d_in[13];
  const float* ln2_g = (const float*)d_in[14]; const float* ln2_b = (const float*)d_in[15];
  const float* W1 = (const float*)d_in[16]; const float* b1 = (const float*)d_in[17];
  const float* W2 = (const float*)d_in[18]; const float* b2 = (const float*)d_in[19];
  const float* fn_g = (const float*)d_in[20]; const float* fn_b = (const float*)d_in[21];
  const float* head_W = (const float*)d_in[22]; const float* head_b = (const float*)d_in[23];

  // ---- workspace layout (~243 MB) ----
  char* ws = (char*)d_ws;
  float*          h   = (float*)ws;                                    //  58,720,256 B
  unsigned short* hn  = (unsigned short*)(ws + 58720256);              //  29,360,128 B
  unsigned short* big = (unsigned short*)(ws + 88080384);              // 117,440,512 B
  unsigned short* qb = big;                                            // q|k|v|o ...
  unsigned short* kb = big + (size_t)ROWS_ * D_;
  unsigned short* vb = big + 2 * (size_t)ROWS_ * D_;
  unsigned short* ob = big + 3 * (size_t)ROWS_ * D_;
  unsigned short* ff = big;                                            // ... or FFN mid (aliased; disjoint lifetimes)
  unsigned short* wT = (unsigned short*)(ws + 88080384 + 117440512);   //  37,748,736 B

  const size_t LSTEP = 3145728;  // bf16 elems of transposed weights per layer

  // ---- weight prep: transpose + bf16 convert (constant work, deterministic) ----
  for (int l = 0; l < NL_; ++l) {
    unsigned short* base = wT + (size_t)l * LSTEP;
    transpose_f32_to_bf16<<<dim3(16, 16), 256, 0, stream>>>(Wq + (size_t)l * D_ * D_, base,            D_, D_);
    transpose_f32_to_bf16<<<dim3(16, 16), 256, 0, stream>>>(Wk + (size_t)l * D_ * D_, base + 262144,   D_, D_);
    transpose_f32_to_bf16<<<dim3(16, 16), 256, 0, stream>>>(Wv + (size_t)l * D_ * D_, base + 524288,   D_, D_);
    transpose_f32_to_bf16<<<dim3(16, 16), 256, 0, stream>>>(Wo + (size_t)l * D_ * D_, base + 786432,   D_, D_);
    transpose_f32_to_bf16<<<dim3(64, 16), 256, 0, stream>>>(W1 + (size_t)l * D_ * DFF_, base + 1048576, D_, DFF_);
    transpose_f32_to_bf16<<<dim3(16, 64), 256, 0, stream>>>(W2 + (size_t)l * DFF_ * D_, base + 2097152, DFF_, D_);
  }

  // ---- patch embedding ----
  patch_embed<<<ROWS_, 256, 0, stream>>>(x, patch_W, patch_b, pos, h);

  // ---- transformer layers ----
  const dim3 gD(ROWS_ / 128, D_ / 128);    // (224, 4)
  const dim3 gF(ROWS_ / 128, DFF_ / 128);  // (224, 16)
  for (int l = 0; l < NL_; ++l) {
    unsigned short* base = wT + (size_t)l * LSTEP;

    layernorm_to_bf16<<<ROWS_ / 8, 256, 0, stream>>>(h, ln1_g + l * D_, ln1_b + l * D_, hn);
    gemm_bf16_wmma<0><<<gD, 256, 0, stream>>>(hn, base,            bq + l * D_, nullptr, qb, ROWS_, D_, D_);
    gemm_bf16_wmma<0><<<gD, 256, 0, stream>>>(hn, base + 262144,   bk + l * D_, nullptr, kb, ROWS_, D_, D_);
    gemm_bf16_wmma<0><<<gD, 256, 0, stream>>>(hn, base + 524288,   bv + l * D_, nullptr, vb, ROWS_, D_, D_);

    attention_wmma<<<B_ * H_, 128, 0, stream>>>(qb, kb, vb, ob);

    gemm_bf16_wmma<1><<<gD, 256, 0, stream>>>(ob, base + 786432,   bo + l * D_, h, nullptr, ROWS_, D_, D_);

    layernorm_to_bf16<<<ROWS_ / 8, 256, 0, stream>>>(h, ln2_g + l * D_, ln2_b + l * D_, hn);
    gemm_bf16_wmma<2><<<gF, 256, 0, stream>>>(hn, base + 1048576, b1 + l * DFF_, nullptr, ff, ROWS_, DFF_, D_);
    gemm_bf16_wmma<1><<<gD, 256, 0, stream>>>(ff, base + 2097152, b2 + l * D_,   h, nullptr, ROWS_, D_, DFF_);
  }

  // ---- pool + final LN + head ----
  pool_ln_head<<<B_, 256, 0, stream>>>(h, fn_g, fn_b, head_W, head_b, (float*)d_out);
}
